// AdaptiveMobiusLayer_19920058319398
// MI455X (gfx1250) — matrix operand order repaired and checked by reading, not compiled, not run on hardware
//
#include <hip/hip_runtime.h>
#include <hip/hip_bf16.h>

typedef __attribute__((ext_vector_type(16))) __bf16 v16bf;
typedef __attribute__((ext_vector_type(8)))  float  v8f;

#define DIM 1024
#define MT  32              // token rows per workgroup
#define THREADS 256         // 8 waves
#define LDA 1032            // padded bf16 row stride: 516 dwords % 64 = 4 -> conflict-free b128

__device__ __forceinline__ float gelu_exact(float v) {
    return 0.5f * v * (1.0f + erff(v * 0.70710678118654752f));
}

// ---------------------------------------------------------------------------
// Repack fp32 weight [K x N] (row-major, k*N+n) into this file's contiguous-k bf16 B-fragment
// layout: block (kt,nt) covers k in [kt*32,kt*32+32), n in [nt*16,nt*16+16).
// Lane L holds n = nt*16 + (L&15), k = kt*32 + (L>=16?16:0)+j, j=0..15
// contiguous -> each lane loads 32 contiguous bytes (two b128).
// ---------------------------------------------------------------------------
__global__ void pack_weights(const float* __restrict__ w, __bf16* __restrict__ p,
                             int K, int N) {
    int o = blockIdx.x * 256 + threadIdx.x;
    if (o >= K * N) return;
    int blk  = o >> 9;
    int rem  = o & 511;
    int lane = rem >> 4;
    int j    = rem & 15;
    int ntiles = N >> 4;
    int kt = blk / ntiles;
    int nt = blk % ntiles;
    int k = kt * 32 + ((lane & 16) ? 16 : 0) + j;
    int n = nt * 16 + (lane & 15);
    p[o] = (__bf16)w[(size_t)k * N + n];
}

// ---------------------------------------------------------------------------
// Column mean over S (two-pass, fixed order => deterministic)
// ---------------------------------------------------------------------------
__global__ void mean_p1(const float* __restrict__ x, float* __restrict__ part) {
    // grid = B(4) * SC(16) * DC(4); block = 256
    int bid = blockIdx.x;
    int dc = bid & 3;
    int sc = (bid >> 2) & 15;
    int b  = bid >> 6;
    int d  = dc * 256 + threadIdx.x;
    const float* xp = x + ((size_t)b * 4096 + (size_t)sc * 256) * DIM + d;
    float s = 0.f;
    for (int i = 0; i < 256; ++i) s += xp[(size_t)i * DIM];
    part[(b * 16 + sc) * DIM + d] = s;
}

__global__ void mean_p2(const float* __restrict__ part, float* __restrict__ mean) {
    int bid = blockIdx.x;                 // B*4
    int dc = bid & 3, b = bid >> 2;
    int d = dc * 256 + threadIdx.x;
    float s = 0.f;
    for (int i = 0; i < 16; ++i) s += part[(b * 16 + i) * DIM + d];
    mean[b * DIM + d] = s * (1.0f / 4096.0f);
}

// ---------------------------------------------------------------------------
// Tiny global-context MLP: mean[b,1024] -> 512 -> 256 -> sigmoid scalar gf[b]
// ---------------------------------------------------------------------------
__global__ __launch_bounds__(256) void gf_kernel(
    const float* __restrict__ mean,
    const float* __restrict__ w1, const float* __restrict__ b1,
    const float* __restrict__ w2, const float* __restrict__ b2,
    const float* __restrict__ w3, const float* __restrict__ b3,
    float* __restrict__ gf) {
    __shared__ float m[1024], h1[512], h2[256], red[256];
    int tid = threadIdx.x;
    for (int b = 0; b < 4; ++b) {
        for (int i = tid; i < 1024; i += 256) m[i] = mean[b * DIM + i];
        __syncthreads();
        for (int n = tid; n < 512; n += 256) {
            float s = b1[n];
            for (int k = 0; k < 1024; ++k) s += m[k] * w1[k * 512 + n];
            h1[n] = gelu_exact(s);
        }
        __syncthreads();
        if (tid < 256) {
            float s = b2[tid];
            for (int k = 0; k < 512; ++k) s += h1[k] * w2[k * 256 + tid];
            h2[tid] = gelu_exact(s);
        }
        __syncthreads();
        red[tid] = h2[tid] * w3[tid];
        __syncthreads();
        for (int off = 128; off > 0; off >>= 1) {
            if (tid < off) red[tid] += red[tid + off];
            __syncthreads();
        }
        if (tid == 0) gf[b] = 1.0f / (1.0f + expf(-(red[0] + b3[0])));
        __syncthreads();
    }
}

// ---------------------------------------------------------------------------
// One GEMM stage on a 32-row tile: dst = gelu(src[32xK] @ W[KxN] + bias)
// src/dst are bf16 in LDS (row stride LDA). W is pre-packed bf16 fragments.
// Each wave owns 4 n-tiles x 2 m-tiles; each B fragment feeds 2 WMMAs.
// Loops are kept NOT unrolled: the body already has 8 independent WMMAs and
// load clauses; unrolling doubles live ranges and caused scratch spills.
// ---------------------------------------------------------------------------
__device__ __forceinline__ void gemm_stage(
    const __bf16* __restrict__ src, __bf16* __restrict__ dst,
    const __bf16* __restrict__ pw, const float* __restrict__ bias,
    int K, int N, int wave, int lane) {
    const int ntiles = N >> 4;
    const int kts    = K >> 5;
    const int nsub   = lane & 15;
    const int mhalf  = lane >> 4;     // A k-base half / C row half

#pragma unroll 1
    for (int ntBase = wave * 4; ntBase < ntiles; ntBase += 32) {
        v8f acc[2][4];
#pragma unroll
        for (int t = 0; t < 4; ++t) {
            float bv = bias[(ntBase + t) * 16 + nsub];
#pragma unroll
            for (int m = 0; m < 2; ++m)
#pragma unroll
                for (int r = 0; r < 8; ++r) acc[m][t][r] = bv;
        }
        // per-kt weight pointer, advanced by ntiles*512 elements per step
        const __bf16* bpk = pw + ((size_t)ntBase << 9) + lane * 16;
        const size_t  bstep = (size_t)ntiles << 9;
#pragma unroll 1
        for (int kt = 0; kt < kts; ++kt) {
            // A fragments: row = m*16 + (lane&15); k = kt*32 + mhalf*16 + {0..15}
            union { v16bf v; uint4 u[2]; } af[2];
#pragma unroll
            for (int m = 0; m < 2; ++m) {
                const __bf16* ap = src + (size_t)(m * 16 + nsub) * LDA
                                       + kt * 32 + mhalf * 16;
                af[m].u[0] = *(const uint4*)ap;
                af[m].u[1] = *(const uint4*)(ap + 8);
            }
#pragma unroll
            for (int t = 0; t < 4; ++t) {
                union { v16bf v; uint4 u[2]; } bfr;
                const __bf16* bp = bpk + ((size_t)t << 9);
                bfr.u[0] = *(const uint4*)bp;
                bfr.u[1] = *(const uint4*)(bp + 8);
                acc[0][t] = __builtin_amdgcn_wmma_f32_16x16x32_bf16(
                    false, af[0].v, false, bfr.v, (short)0, acc[0][t], false, false);
                acc[1][t] = __builtin_amdgcn_wmma_f32_16x16x32_bf16(
                    false, af[1].v, false, bfr.v, (short)0, acc[1][t], false, false);
            }
            bpk += bstep;
        }
        // Epilogue: exact GELU, store bf16 (C layout: row = r + 8*mhalf, col = nsub)
#pragma unroll
        for (int t = 0; t < 4; ++t) {
            int ncol = (ntBase + t) * 16 + nsub;
#pragma unroll
            for (int m = 0; m < 2; ++m)
#pragma unroll
                for (int r = 0; r < 8; ++r) {
                    int row = m * 16 + r + (mhalf << 3);
                    dst[(size_t)row * LDA + ncol] = (__bf16)gelu_exact(acc[m][t][r]);
                }
        }
    }
}

// ---------------------------------------------------------------------------
// Fused 3-cycle Mobius kernel: 32 tokens per workgroup, everything in LDS.
// LDS ~259KB => 1 WG/WGP (2 waves/SIMD): tell the compiler so it can use a
// large VGPR budget instead of spilling.
// ---------------------------------------------------------------------------
__global__ __launch_bounds__(256, 2) void mobius_main(
    const float* __restrict__ x,
    const __bf16* __restrict__ pw1, const float* __restrict__ b1,
    const __bf16* __restrict__ pw2, const float* __restrict__ b2,
    const __bf16* __restrict__ pw3, const float* __restrict__ b3,
    const float* __restrict__ w4, const float* __restrict__ b4,
    const float* __restrict__ gf, const float* __restrict__ arange,
    float* __restrict__ out) {
    extern __shared__ char smem[];
    float*  s_out = (float*)smem;                        // 32*1024 f32 (128KB)
    __bf16* s_a   = (__bf16*)(smem + 131072);            // 32*LDA bf16 (~65KB)
    __bf16* s_b   = (__bf16*)(smem + 131072 + 66560);    // 32*LDA bf16 (~65KB)
    float*  s_red = (float*)(smem + 131072 + 2 * 66560); // 256 f32
    float*  s_tf  = s_red + 256;                         // 32 f32

    const int tid  = threadIdx.x;
    const int wave = tid >> 5;
    const int lane = tid & 31;
    const size_t rowBase = (size_t)blockIdx.x * MT;
    const int batch = (int)(rowBase >> 12);              // / 4096 (32 | 4096)

    const float* xt = x + rowBase * DIM;
    for (int i = tid; i < MT * DIM; i += THREADS) {
        float v = xt[i];
        int row = i >> 10, d = i & 1023;
        s_out[i] = v;
        s_a[(size_t)row * LDA + d] = (__bf16)v;
    }
    __syncthreads();

    const float gfv = gf[batch];
    const float ar  = arange[0];

#pragma unroll 1
    for (int cyc = 0; cyc < 3; ++cyc) {
        gemm_stage(s_a, s_b, pw1, b1, 1024, 1024, wave, lane);  // h1 -> s_b
        __syncthreads();
        gemm_stage(s_b, s_a, pw2, b2, 1024, 512, wave, lane);   // h2 -> s_a
        __syncthreads();
        gemm_stage(s_a, s_b, pw3, b3, 512, 256, wave, lane);    // h3 -> s_b
        __syncthreads();

        // tf = sigmoid(h3 @ w4 + b4), per row: 8 threads x 32 elems per row
        {
            int row = tid >> 3;
            int seg = tid & 7;
            float p = 0.f;
#pragma unroll
            for (int j = 0; j < 32; ++j) {
                int n = seg * 32 + j;
                p += (float)s_b[(size_t)row * LDA + n] * w4[n];
            }
            s_red[tid] = p;
            __syncthreads();
            if (tid < MT) {
                float s = b4[0];
#pragma unroll
                for (int j = 0; j < 8; ++j) s += s_red[tid * 8 + j];
                s_tf[tid] = 1.0f / (1.0f + expf(-s));
            }
            __syncthreads();
        }

        // out = out + coupling * twist(out), pairwise (d, d+512) in fp32
        for (int i = tid; i < MT * 512; i += THREADS) {
            int row = i >> 9;
            int d   = i & 511;
            float comb = 0.7f * gfv + 0.3f * s_tf[row];
            float c = 0.1f + ar * (comb - 0.5f) * 2.0f;
            float a  = s_out[(size_t)row * DIM + d];
            float bb = s_out[(size_t)row * DIM + d + 512];
            float na, nb;
            if (d < 256) { na = a + c * bb; nb = bb - c * a; }   // p_r / -o_r
            else         { na = a - c * bb; nb = bb + c * a; }   // -p_i / o_i
            s_out[(size_t)row * DIM + d]       = na;
            s_out[(size_t)row * DIM + d + 512] = nb;
            if (cyc < 2) {
                s_a[(size_t)row * LDA + d]       = (__bf16)na;
                s_a[(size_t)row * LDA + d + 512] = (__bf16)nb;
            }
        }
        __syncthreads();
    }

    float* ot = out + rowBase * DIM;
    for (int i = tid; i < MT * DIM; i += THREADS) ot[i] = s_out[i];
}

// ---------------------------------------------------------------------------
extern "C" void kernel_launch(void* const* d_in, const int* in_sizes, int n_in,
                              void* d_out, int out_size, void* d_ws, size_t ws_size,
                              hipStream_t stream) {
    const float* x   = (const float*)d_in[0];
    const float* cw1 = (const float*)d_in[1];
    const float* cb1 = (const float*)d_in[2];
    const float* cw2 = (const float*)d_in[3];
    const float* cb2 = (const float*)d_in[4];
    const float* cw3 = (const float*)d_in[5];
    const float* cb3 = (const float*)d_in[6];
    const float* cw4 = (const float*)d_in[7];
    const float* cb4 = (const float*)d_in[8];
    const float* gw1 = (const float*)d_in[9];
    const float* gb1 = (const float*)d_in[10];
    const float* gw2 = (const float*)d_in[11];
    const float* gb2 = (const float*)d_in[12];
    const float* gw3 = (const float*)d_in[13];
    const float* gb3 = (const float*)d_in[14];
    const float* ar  = (const float*)d_in[15];

    char* ws = (char*)d_ws;
    __bf16* pw1 = (__bf16*)ws;            ws += (size_t)1024 * 1024 * 2;
    __bf16* pw2 = (__bf16*)ws;            ws += (size_t)1024 * 512  * 2;
    __bf16* pw3 = (__bf16*)ws;            ws += (size_t)512  * 256  * 2;
    float*  part = (float*)ws;            ws += (size_t)4 * 16 * 1024 * 4;
    float*  mean = (float*)ws;            ws += (size_t)4 * 1024 * 4;
    float*  gfv  = (float*)ws;            ws += 256;

    pack_weights<<<4096, 256, 0, stream>>>(cw1, pw1, 1024, 1024);
    pack_weights<<<2048, 256, 0, stream>>>(cw2, pw2, 1024, 512);
    pack_weights<<<512,  256, 0, stream>>>(cw3, pw3, 512, 256);

    mean_p1<<<256, 256, 0, stream>>>(x, part);
    mean_p2<<<16,  256, 0, stream>>>(part, mean);
    gf_kernel<<<1, 256, 0, stream>>>(mean, gw1, gb1, gw2, gb2, gw3, gb3, gfv);

    size_t shmem = 131072 + 2 * 66560 + 1024 + 128;      // ~259 KB
    mobius_main<<<512, 256, shmem, stream>>>(
        x, pw1, cb1, pw2, cb2, pw3, cb3, cw4, cb4, gfv, ar, (float*)d_out);
}
